// MoEDiscriminator_87806311400124
// MI455X (gfx1250) — matrix operand order repaired and checked by reading, not compiled, not run on hardware
//
#include <hip/hip_runtime.h>
#include <hip/hip_bf16.h>
#include <stdint.h>

typedef __bf16 bf16_t;
typedef __attribute__((ext_vector_type(16))) __bf16 v16bf;
typedef __attribute__((ext_vector_type(8)))  float  v8f;

#define BDIM  65536
#define DS    64
#define NC    8
#define HD    256
#define BM    64      // batch rows per workgroup
#define PITCH 264     // bf16 per LDS row: 256 + 8 pad -> 528B row stride (16B aligned, bank-skewed)

union FragU { uint4 q[2]; v16bf v; };

// A-fragment (16x32 bf16): lane = (m, half); elements 0-15 = K half*16..+15
__device__ inline v16bf load_a_frag(const bf16_t* buf, int mt, int kt, int m, int half) {
  const bf16_t* p = buf + (mt * 16 + m) * PITCH + kt * 32 + half * 16;
  FragU u;
  u.q[0] = *(const uint4*)(p);
  u.q[1] = *(const uint4*)(p + 8);
  return u.v;
}

// B-fragment (32x16 bf16) pre-swizzled: lane L owns 16 contiguous bf16
__device__ inline v16bf load_b_frag(const bf16_t* w, int fragIdx, int lane) {
  const bf16_t* p = w + (size_t)fragIdx * 512 + lane * 16;
  FragU u;
  u.q[0] = *(const uint4*)(p);
  u.q[1] = *(const uint4*)(p + 8);
  return u.v;
}

// Pre-swizzle W1 [C][64][256] and W2 [C][256][256] (f32) into bf16 B-fragment layout.
// Fragment (c, kt, nt): lane L = (n = L&15, khalf = L>>4) holds W[kt*32 + khalf*16 + j][nt*16 + n], j=0..15.
__global__ __launch_bounds__(256) void swizzle_weights(const float* __restrict__ W1,
                                                       const float* __restrict__ W2,
                                                       bf16_t* __restrict__ w1s,
                                                       bf16_t* __restrict__ w2s) {
  int t = blockIdx.x * 256 + threadIdx.x;      // one thread per (fragment, lane)
  int fl = t >> 5, lane = t & 31;
  const int NFRAG1 = NC * 2 * 16;              // 256 fragments (K=64 -> 2 k-tiles)
  const int NFRAG2 = NC * 8 * 16;              // 1024 fragments (K=256 -> 8 k-tiles)
  int n_in_tile = lane & 15;
  int khalf = lane >> 4;
  if (fl < NFRAG1) {
    int c = fl / 32, f = fl % 32, kt = f / 16, nt = f % 16;
    const float* W = W1 + (size_t)c * DS * HD;
    bf16_t* dst = w1s + (size_t)fl * 512 + lane * 16;
    int n = nt * 16 + n_in_tile;
    int k0 = kt * 32 + khalf * 16;
    #pragma unroll
    for (int j = 0; j < 16; ++j) dst[j] = (bf16_t)W[(size_t)(k0 + j) * HD + n];
  } else if (fl < NFRAG1 + NFRAG2) {
    int g = fl - NFRAG1;
    int c = g / 128, f = g % 128, kt = f / 16, nt = f % 16;
    const float* W = W2 + (size_t)c * HD * HD;
    bf16_t* dst = w2s + (size_t)g * 512 + lane * 16;
    int n = nt * 16 + n_in_tile;
    int k0 = kt * 32 + khalf * 16;
    #pragma unroll
    for (int j = 0; j < 16; ++j) dst[j] = (bf16_t)W[(size_t)(k0 + j) * HD + n];
  }
}

__global__ __launch_bounds__(256) void moe_disc_fused(
    const float*  __restrict__ st,
    const bf16_t* __restrict__ w1s, const float* __restrict__ b1,
    const bf16_t* __restrict__ w2s, const float* __restrict__ b2,
    const float*  __restrict__ W3,  const float* __restrict__ b3,
    float* __restrict__ out) {
  extern __shared__ bf16_t smem[];
  bf16_t* buf0 = smem;                 // st tile, later h2
  bf16_t* buf1 = smem + BM * PITCH;    // h1, later f32 reduction scratch

  const int c   = blockIdx.y;
  const int b0  = blockIdx.x * BM;
  const int tid = threadIdx.x;
  const int w    = tid >> 5;
  const int lane = tid & 31;
  const int m    = lane & 15;
  const int half = lane >> 4;
  const int ns   = w * 32;             // this wave's N-slice base

  // ---- Stage st tile [BM x 64] f32 -> bf16 into buf0 ----
  {
    int row = tid >> 2;
    int seg = (tid & 3) * 16;
    const float4* src = (const float4*)(st + (size_t)(b0 + row) * DS + seg);
    bf16_t* dst = buf0 + row * PITCH + seg;
    #pragma unroll
    for (int i = 0; i < 4; ++i) {
      float4 f = src[i];
      dst[i * 4 + 0] = (bf16_t)f.x;
      dst[i * 4 + 1] = (bf16_t)f.y;
      dst[i * 4 + 2] = (bf16_t)f.z;
      dst[i * 4 + 3] = (bf16_t)f.w;
    }
  }
  __syncthreads();

  const v8f vzero = {0.f, 0.f, 0.f, 0.f, 0.f, 0.f, 0.f, 0.f};

  // ---- Layer 1: h1 = relu(st @ W1[c] + b1[c])  [64x64]x[64x256] ----
  {
    v8f acc[4][2];
    #pragma unroll
    for (int mt = 0; mt < 4; ++mt)
      #pragma unroll
      for (int nt = 0; nt < 2; ++nt) acc[mt][nt] = vzero;

    #pragma unroll
    for (int kt = 0; kt < 2; ++kt) {
      v16bf a[4];
      #pragma unroll
      for (int mt = 0; mt < 4; ++mt) a[mt] = load_a_frag(buf0, mt, kt, m, half);
      v16bf bm[2];
      #pragma unroll
      for (int nt = 0; nt < 2; ++nt)
        bm[nt] = load_b_frag(w1s, c * 32 + kt * 16 + (w * 2 + nt), lane);
      #pragma unroll
      for (int mt = 0; mt < 4; ++mt)
        #pragma unroll
        for (int nt = 0; nt < 2; ++nt)
          acc[mt][nt] = __builtin_amdgcn_wmma_f32_16x16x32_bf16(
              false, a[mt], false, bm[nt], (short)0, acc[mt][nt], false, false);
    }
    // bias + relu + bf16 -> buf1 (row-major)
    #pragma unroll
    for (int nt = 0; nt < 2; ++nt) {
      int col = ns + nt * 16 + m;
      float bias = b1[c * HD + col];
      #pragma unroll
      for (int mt = 0; mt < 4; ++mt) {
        #pragma unroll
        for (int v = 0; v < 8; ++v) {
          int row = mt * 16 + half * 8 + v;
          float x = acc[mt][nt][v] + bias;
          buf1[row * PITCH + col] = (bf16_t)fmaxf(x, 0.f);
        }
      }
    }
  }
  __syncthreads();

  // ---- Layer 2: h2 = relu(h1 @ W2[c] + b2[c])  [64x256]x[256x256] ----
  {
    v8f acc[4][2];
    #pragma unroll
    for (int mt = 0; mt < 4; ++mt)
      #pragma unroll
      for (int nt = 0; nt < 2; ++nt) acc[mt][nt] = vzero;

    for (int kt = 0; kt < 8; ++kt) {
      v16bf a[4];
      #pragma unroll
      for (int mt = 0; mt < 4; ++mt) a[mt] = load_a_frag(buf1, mt, kt, m, half);
      v16bf bm[2];
      #pragma unroll
      for (int nt = 0; nt < 2; ++nt)
        bm[nt] = load_b_frag(w2s, c * 128 + kt * 16 + (w * 2 + nt), lane);
      #pragma unroll
      for (int mt = 0; mt < 4; ++mt)
        #pragma unroll
        for (int nt = 0; nt < 2; ++nt)
          acc[mt][nt] = __builtin_amdgcn_wmma_f32_16x16x32_bf16(
              false, a[mt], false, bm[nt], (short)0, acc[mt][nt], false, false);
    }
    #pragma unroll
    for (int nt = 0; nt < 2; ++nt) {
      int col = ns + nt * 16 + m;
      float bias = b2[c * HD + col];
      #pragma unroll
      for (int mt = 0; mt < 4; ++mt) {
        #pragma unroll
        for (int v = 0; v < 8; ++v) {
          int row = mt * 16 + half * 8 + v;
          float x = acc[mt][nt][v] + bias;
          buf0[row * PITCH + col] = (bf16_t)fmaxf(x, 0.f);
        }
      }
    }
  }
  __syncthreads();

  // ---- Layer 3: d = h2 @ W3[c] + b3[c]  (N=1 -> VALU dot, 0.4% of FLOPs) ----
  {
    int row  = tid & 63;
    int part = tid >> 6;                 // 4 partial sums of 64 K each
    const float* w3 = W3 + (size_t)c * HD;
    float acc = 0.f;
    int k0 = part * 64;
    #pragma unroll 8
    for (int k = 0; k < 64; ++k)
      acc += (float)buf0[row * PITCH + k0 + k] * w3[k0 + k];
    float* red = (float*)buf1;           // h1 no longer needed
    red[tid] = acc;
    __syncthreads();
    if (tid < 64) {
      float s = red[tid] + red[64 + tid] + red[128 + tid] + red[192 + tid];
      out[(size_t)(b0 + tid) * NC + c] = s + b3[c];
    }
  }
}

extern "C" void kernel_launch(void* const* d_in, const int* in_sizes, int n_in,
                              void* d_out, int out_size, void* d_ws, size_t ws_size,
                              hipStream_t stream) {
  (void)in_sizes; (void)n_in; (void)out_size; (void)ws_size;
  const float* st = (const float*)d_in[0];
  const float* W1 = (const float*)d_in[1];
  const float* b1 = (const float*)d_in[2];
  const float* W2 = (const float*)d_in[3];
  const float* b2 = (const float*)d_in[4];
  const float* W3 = (const float*)d_in[5];
  const float* b3 = (const float*)d_in[6];
  float* out = (float*)d_out;

  // ws: [0, 256KiB) = W1 swizzled bf16; [256KiB, 1.25MiB) = W2 swizzled bf16
  bf16_t* w1s = (bf16_t*)d_ws;
  bf16_t* w2s = (bf16_t*)((char*)d_ws + (size_t)NC * 2 * 16 * 512 * sizeof(bf16_t));

  const int nfrag = NC * 2 * 16 + NC * 8 * 16;           // 1280 fragments
  hipLaunchKernelGGL(swizzle_weights, dim3((nfrag * 32) / 256), dim3(256), 0, stream,
                     W1, W2, w1s, w2s);

  size_t shbytes = (size_t)2 * BM * PITCH * sizeof(bf16_t);   // 67584 B
  hipFuncSetAttribute(reinterpret_cast<const void*>(moe_disc_fused),
                      hipFuncAttributeMaxDynamicSharedMemorySize, (int)shbytes);
  hipLaunchKernelGGL(moe_disc_fused, dim3(BDIM / BM, NC), dim3(256), shbytes, stream,
                     st, w1s, b1, w2s, b2, W3, b3, out);
}
